// DoubleSwinTransformerBlock_37323265802398
// MI455X (gfx1250) — matrix operand order repaired and checked by reading, not compiled, not run on hardware
//
#include <hip/hip_runtime.h>
#include <hip/hip_bf16.h>

typedef __attribute__((ext_vector_type(16))) _Float16 v16h;
typedef __attribute__((ext_vector_type(8)))  _Float16 v8h;
typedef __attribute__((ext_vector_type(8)))  float    v8f;

#define WMMA16(a, b, c) \
  __builtin_amdgcn_wmma_f32_16x16x32_f16(false, (a), false, (b), (short)0, (c), false, false)

// gfx1250 async global->LDS path (ASYNCcnt), guarded so compile never breaks.
// Probe (round 2 diagnostic) showed the builtin takes v4i pointers:
//   (v4i addrspace(1)*, v4i addrspace(3)*, imm offset, imm cpol)
#if __has_builtin(__builtin_amdgcn_global_load_async_to_lds_b128) && \
    __has_builtin(__builtin_amdgcn_s_wait_asynccnt)
#define HAVE_ASYNC_LDS 1
typedef int v4i_vs __attribute__((__vector_size__(16)));
typedef __attribute__((address_space(1))) v4i_vs* gb128_t;
typedef __attribute__((address_space(3))) v4i_vs* sb128_t;
#else
#define HAVE_ASYNC_LDS 0
#endif

// ---------------------------------------------------------------------------
// Fragment helpers (layouts per CDNA5 ISA, wave32)
// A (16x32 f16): lanes 0-15 -> M=lane, K = {kb..kb+7, kb+16..kb+23} kb=0
//                lanes 16-31 -> same M set, kb=8
// ---------------------------------------------------------------------------
__device__ __forceinline__ v16h frag_a(const _Float16* tile, int ld, int lane) {
  const int m  = lane & 15;
  const int kb = (lane & 16) ? 8 : 0;
  const _Float16* p = tile + m * ld + kb;
  v8h lo = *(const v8h*)(p);
  v8h hi = *(const v8h*)(p + 16);
  v16h r;
#pragma unroll
  for (int i = 0; i < 8; ++i) { r[i] = lo[i]; r[i + 8] = hi[i]; }
  return r;
}

// B (32x16 f16): lane n<16 -> column N=n, K={0..7,16..23} ; lane>=16 -> N=n-16, K={8..15,24..31}
// w points at W[n0][k0] of a row-major (N,K) matrix -> two 8-half loads.
__device__ __forceinline__ v16h frag_b(const _Float16* w, int ld, int lane) {
  const int n  = lane & 15;
  const int kb = (lane & 16) ? 8 : 0;
  const _Float16* p = w + (size_t)n * ld + kb;
  v8h lo = *(const v8h*)(p);
  v8h hi = *(const v8h*)(p + 16);
  v16h r;
#pragma unroll
  for (int i = 0; i < 8; ++i) { r[i] = lo[i]; r[i + 8] = hi[i]; }
  return r;
}

__device__ __forceinline__ float rmax16(float v) {
  v = fmaxf(v, __shfl_xor(v, 1, 32));
  v = fmaxf(v, __shfl_xor(v, 2, 32));
  v = fmaxf(v, __shfl_xor(v, 4, 32));
  v = fmaxf(v, __shfl_xor(v, 8, 32));
  return v;
}
__device__ __forceinline__ float rsum16(float v) {
  v += __shfl_xor(v, 1, 32);
  v += __shfl_xor(v, 2, 32);
  v += __shfl_xor(v, 4, 32);
  v += __shfl_xor(v, 8, 32);
  return v;
}
__device__ __forceinline__ float rsum32(float v) {
  v += __shfl_xor(v, 16, 32);
  return rsum16(v);
}

__device__ __forceinline__ int regid1(int t) {  // Swin shift region id, H=224 ws=7 shift=3
  return t < 217 ? 0 : (t < 221 ? 1 : 2);
}

// ---------------------------------------------------------------------------
// fp32 -> f16 weight conversion
// ---------------------------------------------------------------------------
__global__ void cvt_f16_kernel(const float* __restrict__ s, _Float16* __restrict__ d, int n) {
  int i = blockIdx.x * 256 + threadIdx.x;
  if (i < n) d[i] = (_Float16)s[i];
}

// ---------------------------------------------------------------------------
// LayerNorm (C=256) -> f16 ; one wave per token, 8 channels per lane
// ---------------------------------------------------------------------------
__global__ void ln1_kernel(const float* __restrict__ x, const float* __restrict__ g,
                           const float* __restrict__ bt, _Float16* __restrict__ out) {
  const int lane  = threadIdx.x & 31;
  const int token = blockIdx.x * 8 + (threadIdx.x >> 5);
  const size_t base = (size_t)token * 256 + lane * 8;
  float v[8];
  float s = 0.f;
#pragma unroll
  for (int i = 0; i < 8; ++i) { v[i] = x[base + i]; s += v[i]; }
  s = rsum32(s);
  const float mu = s * (1.0f / 256.0f);
  float q = 0.f;
#pragma unroll
  for (int i = 0; i < 8; ++i) { float d = v[i] - mu; q += d * d; }
  q = rsum32(q);
  const float rstd = rsqrtf(q * (1.0f / 256.0f) + 1e-5f);
  const int c0 = lane * 8;
#pragma unroll
  for (int i = 0; i < 8; ++i)
    out[base + i] = (_Float16)((v[i] - mu) * rstd * g[c0 + i] + bt[c0 + i]);
}

// residual add + LN2 ; writes x2 (fp32, into d_out) and normalized f16
__global__ void add_ln2_kernel(const float* __restrict__ x, const float* __restrict__ y,
                               const float* __restrict__ g, const float* __restrict__ bt,
                               float* __restrict__ x2, _Float16* __restrict__ out) {
  const int lane  = threadIdx.x & 31;
  const int token = blockIdx.x * 8 + (threadIdx.x >> 5);
  const size_t base = (size_t)token * 256 + lane * 8;
  float v[8];
  float s = 0.f;
#pragma unroll
  for (int i = 0; i < 8; ++i) { v[i] = x[base + i] + y[base + i]; s += v[i]; }
  s = rsum32(s);
  const float mu = s * (1.0f / 256.0f);
  float q = 0.f;
#pragma unroll
  for (int i = 0; i < 8; ++i) { float d = v[i] - mu; q += d * d; }
  q = rsum32(q);
  const float rstd = rsqrtf(q * (1.0f / 256.0f) + 1e-5f);
  const int c0 = lane * 8;
#pragma unroll
  for (int i = 0; i < 8; ++i) {
    x2[base + i]  = v[i];
    out[base + i] = (_Float16)((v[i] - mu) * rstd * g[c0 + i] + bt[c0 + i]);
  }
}

// ---------------------------------------------------------------------------
// Fused shifted-window attention for one half (128 ch, 4 heads x hd=32).
// One block per window (1024 windows), blockIdx.y = batch, one wave per head.
// LDS: Xs 64x128 | per-head {Qs 64x32, Ks 64x32, Vt 32x64, As 16x64} ; Os aliases Xs.
// ---------------------------------------------------------------------------
#define ATTN_LDS_BYTES ((64 * 128 + 4 * (64 * 32 + 64 * 32 + 32 * 64 + 16 * 64)) * 2)

__global__ void swin_attn_kernel(const _Float16* __restrict__ xn,
                                 const _Float16* __restrict__ qkvw,
                                 const float*    __restrict__ qkvb,
                                 const _Float16* __restrict__ projw,
                                 const float*    __restrict__ projb,
                                 const float*    __restrict__ rpb,
                                 float*          __restrict__ y,
                                 int hsel) {
  extern __shared__ _Float16 smem[];
  const int tid  = threadIdx.x;
  const int lane = tid & 31;
  const int h    = tid >> 5;         // head 0..3
  const int widx = blockIdx.x;       // 0..1023
  const int wh   = widx >> 5;
  const int wwi  = widx & 31;
  const int b    = blockIdx.y;

  _Float16* Xs = smem;                                     // 64 x 128
  _Float16* hb = smem + 64 * 128 + h * (64 * 32 + 64 * 32 + 32 * 64 + 16 * 64);
  _Float16* Qs = hb;                                       // 64 x 32
  _Float16* Ks = hb + 64 * 32;                             // 64 x 32 (row=token, col=d)
  _Float16* Vt = hb + 2 * 64 * 32;                         // 32 x 64 (row=d, col=token)
  _Float16* As = hb + 2 * 64 * 32 + 32 * 64;               // 16 x 64
  _Float16* Os = smem;                                     // alias Xs: 64 x 128

  // ---- stage shifted-window X tile into LDS (rows 49..63 zero) ----
  for (int idx = tid; idx < 64 * 16; idx += 128) {
    const int row = idx >> 4, c8 = idx & 15;
    if (row < 49) {
      const int r  = row / 7, c = row - r * 7;
      const int gh = (wh * 7 + r + 3) % 224;     // roll(-3)
      const int gw = (wwi * 7 + c + 3) % 224;
      const _Float16* src =
          xn + (((size_t)b * 224 + gh) * 224 + gw) * 256 + hsel * 128 + c8 * 8;
#if HAVE_ASYNC_LDS
      __builtin_amdgcn_global_load_async_to_lds_b128(
          (gb128_t)src, (sb128_t)(Xs + row * 128 + c8 * 8), 0, 0);
#else
      *(v8h*)(Xs + row * 128 + c8 * 8) = *(const v8h*)src;
#endif
    } else {
      v8h z;
#pragma unroll
      for (int i = 0; i < 8; ++i) z[i] = (_Float16)0.f;
      *(v8h*)(Xs + row * 128 + c8 * 8) = z;
    }
  }
#if HAVE_ASYNC_LDS
  __builtin_amdgcn_s_wait_asynccnt(0);
#endif
  __syncthreads();

  const int nlo = lane & 15;
  const int mo  = (lane & 16) ? 8 : 0;
  const float qscale = 0.17677669529663687f;  // 32^-0.5

  // ---- QKV projection: X(64x128) @ Wp^T(128x32) per head ----
  for (int pj = 0; pj < 3; ++pj) {
    const int rowbase = pj * 128 + h * 32;
    for (int mt = 0; mt < 4; ++mt) {
      v8f c[2] = {};
#pragma unroll
      for (int kt = 0; kt < 4; ++kt) {
        v16h a = frag_a(Xs + mt * 16 * 128 + kt * 32, 128, lane);
#pragma unroll
        for (int nt = 0; nt < 2; ++nt) {
          v16h bf = frag_b(qkvw + (size_t)(rowbase + nt * 16) * 128 + kt * 32, 128, lane);
          c[nt] = WMMA16(a, bf, c[nt]);
        }
      }
#pragma unroll
      for (int nt = 0; nt < 2; ++nt) {
        const float bias = qkvb[rowbase + nt * 16 + nlo];
        if (pj == 0) {
          _Float16* dst = Qs + mt * 16 * 32 + nt * 16;
#pragma unroll
          for (int v = 0; v < 8; ++v) dst[(mo + v) * 32 + nlo] = (_Float16)((c[nt][v] + bias) * qscale);
        } else if (pj == 1) {
          _Float16* dst = Ks + mt * 16 * 32 + nt * 16;
#pragma unroll
          for (int v = 0; v < 8; ++v) dst[(mo + v) * 32 + nlo] = (_Float16)(c[nt][v] + bias);
        } else {  // V stored transposed: Vt[d][token]
          _Float16* dst = Vt + (nt * 16) * 64 + mt * 16;
#pragma unroll
          for (int v = 0; v < 8; ++v) dst[nlo * 64 + mo + v] = (_Float16)(c[nt][v] + bias);
        }
      }
    }
  }
  __syncthreads();  // also protects Os aliasing Xs

  // ---- attention: per m-tile QK^T, bias+mask+softmax, AV ----
  for (int mt = 0; mt < 4; ++mt) {
    v8f acc[4] = {};
    v16h aq = frag_a(Qs + mt * 16 * 32, 32, lane);
#pragma unroll
    for (int nt = 0; nt < 4; ++nt) {
      v16h bk = frag_b(Ks + nt * 16 * 32, 32, lane);  // B[k=d][n=token]: Ks row-major works
      acc[nt] = WMMA16(aq, bk, acc[nt]);
    }
#pragma unroll
    for (int v = 0; v < 8; ++v) {
      const int i  = mt * 16 + mo + v;
      const int ri = i / 7, ci = i - ri * 7;
      float l[4];
#pragma unroll
      for (int nt = 0; nt < 4; ++nt) {
        const int j = nt * 16 + nlo;
        float lv = -1e30f;
        if (i < 49 && j < 49) {
          const int rj = j / 7, cj = j - rj * 7;
          const float bias = rpb[(size_t)((ri - rj + 6) * 13 + (ci - cj + 6)) * 4 + h];
          const int idi = regid1(wh * 7 + ri) * 3 + regid1(wwi * 7 + ci);
          const int idj = regid1(wh * 7 + rj) * 3 + regid1(wwi * 7 + cj);
          float msk = (idi != idj) ? -100.f : 0.f;
          const float dy = (float)(ri - rj) * 20.0f;
          const float dt = (float)(ci - cj);
          bool allow;
          if (i == j) allow = true;
          else if (dt > 0.f) {
            if (hsel == 0)
              allow = (dy >= 0.f) && (dy >= 16.666666f * dt) && (dy <= 27.777779f * dt);
            else {
              const float ady = fabsf(dy);
              allow = (dy <= 0.f) && (ady >= 2.7777777f * dt) && (ady <= 6.9444447f * dt);
            }
          } else allow = false;
          msk += allow ? 0.f : -100.f;
          lv = acc[nt][v] + bias + msk;
        }
        l[nt] = lv;
      }
      float lmax = fmaxf(fmaxf(l[0], l[1]), fmaxf(l[2], l[3]));
      lmax = rmax16(lmax);
      float lsum = 0.f;
#pragma unroll
      for (int nt = 0; nt < 4; ++nt) { float e = __expf(l[nt] - lmax); l[nt] = e; lsum += e; }
      lsum = rsum16(lsum);
      const float inv = 1.0f / lsum;
#pragma unroll
      for (int nt = 0; nt < 4; ++nt)
        As[(mo + v) * 64 + nt * 16 + nlo] = (_Float16)(l[nt] * inv);
    }
    // AV: As(16x64) @ V(64x32), V staged transposed so B loads are contiguous
    v8f o[2] = {};
#pragma unroll
    for (int kt = 0; kt < 2; ++kt) {
      v16h aa = frag_a(As + kt * 32, 64, lane);
#pragma unroll
      for (int nt = 0; nt < 2; ++nt) {
        v16h bv = frag_b(Vt + (nt * 16) * 64 + kt * 32, 64, lane);
        o[nt] = WMMA16(aa, bv, o[nt]);
      }
    }
#pragma unroll
    for (int nt = 0; nt < 2; ++nt)
#pragma unroll
      for (int v = 0; v < 8; ++v)
        Os[(mt * 16 + mo + v) * 128 + h * 32 + nt * 16 + nlo] = (_Float16)o[nt][v];
  }
  __syncthreads();

  // ---- output projection Os(64x128) @ Wp^T(128x32-slice), scatter with roll(+3) ----
  for (int mt = 0; mt < 4; ++mt) {
    v8f c[2] = {};
#pragma unroll
    for (int kt = 0; kt < 4; ++kt) {
      v16h a = frag_a(Os + mt * 16 * 128 + kt * 32, 128, lane);
#pragma unroll
      for (int nt = 0; nt < 2; ++nt) {
        v16h bf = frag_b(projw + (size_t)(h * 32 + nt * 16) * 128 + kt * 32, 128, lane);
        c[nt] = WMMA16(a, bf, c[nt]);
      }
    }
#pragma unroll
    for (int nt = 0; nt < 2; ++nt) {
      const int col = h * 32 + nt * 16 + nlo;
      const float bias = projb[col];
#pragma unroll
      for (int v = 0; v < 8; ++v) {
        const int tok = mt * 16 + mo + v;
        if (tok < 49) {
          const int r  = tok / 7, c2 = tok - r * 7;
          const int gh = (wh * 7 + r + 3) % 224;
          const int gw = (wwi * 7 + c2 + 3) % 224;
          y[(((size_t)b * 224 + gh) * 224 + gw) * 256 + hsel * 128 + col] = c[nt][v] + bias;
        }
      }
    }
  }
}

// ---------------------------------------------------------------------------
// MLP GEMMs: per-wave 64x32 output tile. acc[4][2] held across the K loop so
// each B fragment feeds 4 WMMAs (4x fewer weight loads than per-mt looping).
// Weights stream through L2 (192 MB >> all operands); prefetch next k-step.
// ---------------------------------------------------------------------------
__global__ void fc1_gelu_kernel(const _Float16* __restrict__ hn, const _Float16* __restrict__ w,
                                const float* __restrict__ bias, _Float16* __restrict__ out) {
  const int lane = threadIdx.x & 31;
  const int wv   = threadIdx.x >> 5;
  const int m0   = blockIdx.x * 64;
  const int n0   = blockIdx.y * 128 + wv * 32;
  const int nlo  = lane & 15, mo = (lane & 16) ? 8 : 0;
  v8f acc[4][2] = {};
  for (int kt = 0; kt < 8; ++kt) {  // K=256
    v16h a[4];
#pragma unroll
    for (int mt = 0; mt < 4; ++mt)
      a[mt] = frag_a(hn + (size_t)(m0 + mt * 16) * 256 + kt * 32, 256, lane);
#pragma unroll
    for (int nt = 0; nt < 2; ++nt) {
      const _Float16* wp = w + (size_t)(n0 + nt * 16) * 256 + kt * 32;
      v16h bf = frag_b(wp, 256, lane);
      if (kt + 1 < 8) __builtin_prefetch(wp + 32, 0, 2);
#pragma unroll
      for (int mt = 0; mt < 4; ++mt)
        acc[mt][nt] = WMMA16(a[mt], bf, acc[mt][nt]);
    }
  }
#pragma unroll
  for (int mt = 0; mt < 4; ++mt)
#pragma unroll
    for (int nt = 0; nt < 2; ++nt) {
      const int col = n0 + nt * 16 + nlo;
      const float bb = bias[col];
#pragma unroll
      for (int v = 0; v < 8; ++v) {
        const float x = acc[mt][nt][v] + bb;
        const float g = 0.5f * x * (1.0f + erff(x * 0.70710678f));  // exact gelu
        out[(size_t)(m0 + mt * 16 + mo + v) * 1024 + col] = (_Float16)g;
      }
    }
}

__global__ void fc2_add_kernel(const _Float16* __restrict__ hm, const _Float16* __restrict__ w,
                               const float* __restrict__ bias, float* __restrict__ out) {
  const int lane = threadIdx.x & 31;
  const int wv   = threadIdx.x >> 5;
  const int m0   = blockIdx.x * 64;
  const int n0   = blockIdx.y * 128 + wv * 32;
  const int nlo  = lane & 15, mo = (lane & 16) ? 8 : 0;
  v8f acc[4][2] = {};
  for (int kt = 0; kt < 32; ++kt) {  // K=1024
    v16h a[4];
#pragma unroll
    for (int mt = 0; mt < 4; ++mt)
      a[mt] = frag_a(hm + (size_t)(m0 + mt * 16) * 1024 + kt * 32, 1024, lane);
#pragma unroll
    for (int nt = 0; nt < 2; ++nt) {
      const _Float16* wp = w + (size_t)(n0 + nt * 16) * 1024 + kt * 32;
      v16h bf = frag_b(wp, 1024, lane);
      if (kt + 1 < 32) __builtin_prefetch(wp + 32, 0, 2);
#pragma unroll
      for (int mt = 0; mt < 4; ++mt)
        acc[mt][nt] = WMMA16(a[mt], bf, acc[mt][nt]);
    }
  }
#pragma unroll
  for (int mt = 0; mt < 4; ++mt)
#pragma unroll
    for (int nt = 0; nt < 2; ++nt) {
      const int col = n0 + nt * 16 + nlo;
      const float bb = bias[col];
#pragma unroll
      for (int v = 0; v < 8; ++v) {
        const size_t o = (size_t)(m0 + mt * 16 + mo + v) * 256 + col;
        out[o] = out[o] + acc[mt][nt][v] + bb;  // out already holds x2 (residual)
      }
    }
}

// ---------------------------------------------------------------------------
extern "C" void kernel_launch(void* const* d_in, const int* in_sizes, int n_in,
                              void* d_out, int out_size, void* d_ws, size_t ws_size,
                              hipStream_t stream) {
  (void)in_sizes; (void)n_in; (void)out_size; (void)ws_size;
  const float* x     = (const float*)d_in[0];
  const float* n1g   = (const float*)d_in[3];
  const float* n1b   = (const float*)d_in[4];
  const float* n2g   = (const float*)d_in[5];
  const float* n2b   = (const float*)d_in[6];
  const float* qw1   = (const float*)d_in[7];
  const float* qb1   = (const float*)d_in[8];
  const float* pw1   = (const float*)d_in[9];
  const float* pb1   = (const float*)d_in[10];
  const float* rpb1  = (const float*)d_in[11];
  const float* qw2   = (const float*)d_in[12];
  const float* qb2   = (const float*)d_in[13];
  const float* pw2   = (const float*)d_in[14];
  const float* pb2   = (const float*)d_in[15];
  const float* rpb2  = (const float*)d_in[16];
  const float* f1w   = (const float*)d_in[17];
  const float* f1b   = (const float*)d_in[18];
  const float* f2w   = (const float*)d_in[19];
  const float* f2b   = (const float*)d_in[20];
  float* out = (float*)d_out;

  const size_t NTOK = (size_t)2 * 224 * 224;  // 100352
  char* p = (char*)d_ws;
  _Float16* xn   = (_Float16*)p; p += NTOK * 256 * 2;
  float*    ybuf = (float*)p;    p += NTOK * 256 * 4;
  _Float16* hn   = (_Float16*)p; p += NTOK * 256 * 2;
  _Float16* hmid = (_Float16*)p; p += NTOK * 1024 * 2;
  _Float16* qw1h = (_Float16*)p; p += 384 * 128 * 2;
  _Float16* qw2h = (_Float16*)p; p += 384 * 128 * 2;
  _Float16* pw1h = (_Float16*)p; p += 128 * 128 * 2;
  _Float16* pw2h = (_Float16*)p; p += 128 * 128 * 2;
  _Float16* f1wh = (_Float16*)p; p += 1024 * 256 * 2;
  _Float16* f2wh = (_Float16*)p; p += 256 * 1024 * 2;

  auto cvt = [&](const float* s, _Float16* d, int n) {
    cvt_f16_kernel<<<(n + 255) / 256, 256, 0, stream>>>(s, d, n);
  };
  cvt(qw1, qw1h, 384 * 128);
  cvt(qw2, qw2h, 384 * 128);
  cvt(pw1, pw1h, 128 * 128);
  cvt(pw2, pw2h, 128 * 128);
  cvt(f1w, f1wh, 1024 * 256);
  cvt(f2w, f2wh, 256 * 1024);

  ln1_kernel<<<NTOK / 8, 256, 0, stream>>>(x, n1g, n1b, xn);

  swin_attn_kernel<<<dim3(1024, 2), 128, ATTN_LDS_BYTES, stream>>>(
      xn, qw1h, qb1, pw1h, pb1, rpb1, ybuf, 0);
  swin_attn_kernel<<<dim3(1024, 2), 128, ATTN_LDS_BYTES, stream>>>(
      xn, qw2h, qb2, pw2h, pb2, rpb2, ybuf, 1);

  add_ln2_kernel<<<NTOK / 8, 256, 0, stream>>>(x, ybuf, n2g, n2b, out, hn);

  fc1_gelu_kernel<<<dim3(NTOK / 64, 8), 128, 0, stream>>>(hn, f1wh, f1b, hmid);
  fc2_add_kernel<<<dim3(NTOK / 64, 2), 128, 0, stream>>>(hmid, f2wh, f2b, out);
}
